// MyModel_61933428412956
// MI455X (gfx1250) — matrix operand order repaired and checked by reading, not compiled, not run on hardware
//
#include <hip/hip_runtime.h>

typedef __attribute__((ext_vector_type(16))) _Float16 v16h;
typedef __attribute__((ext_vector_type(8)))  _Float16 v8h;
typedef __attribute__((ext_vector_type(8)))  float    v8f;

#define NB   4096
#define TT   512
#define DIN  32
#define HH   16

__device__ __forceinline__ float sigmoid_fast(float x) {
    return 1.0f / (1.0f + __expf(-x));
}
__device__ __forceinline__ float tanh_fast(float x) {
    float e = __expf(2.0f * x);
    return (e - 1.0f) / (e + 1.0f);
}

// One wave (32 lanes) owns 16 batch rows; both LSTM layers advance together per
// timestep. Layer 0: gates = bias + x_t(16x32)*Wih0^T + h1(16x16,padK)*Whh0^T
// (2 chained WMMAs per gate tile). Layer 1 packs its two K=16 matmuls into ONE
// K=32 WMMA per gate tile: A = [h1_new | h2_prev], B = [Wih1^T ; Whh1^T].
// 12 v_wmma_f32_16x16x32_f16 per timestep.
__global__ __launch_bounds__(32)
void lstm2_fused_kernel(const float* __restrict__ x,
                        const float* __restrict__ Wih0, const float* __restrict__ Whh0,
                        const float* __restrict__ bih0, const float* __restrict__ bhh0,
                        const float* __restrict__ Wih1, const float* __restrict__ Whh1,
                        const float* __restrict__ bih1, const float* __restrict__ bhh1,
                        const float* __restrict__ fc1w, const float* __restrict__ fc1b,
                        const float* __restrict__ fc2w, const float* __restrict__ fc2b,
                        float* __restrict__ out)
{
    __shared__ alignas(16) _Float16 lds_h1[16 * 16];   // row-major [m][n], f16
    __shared__ alignas(16) _Float16 lds_h2[16 * 16];
    __shared__ alignas(16) float    lds_last[16 * 16];

    const int lane = threadIdx.x & 31;
    const int half = lane >> 4;       // which K half this lane supplies
    const int lm   = lane & 15;
    const int b0   = blockIdx.x * 16; // batch tile base

    // ---------------- preload weight B-operands into registers ----------------
    // Dense 16-bit B (KxN = 32x16) layout assumption (mirrors documented 8-bit B):
    //   lane l holds column N = l%16; element j holds K = 8*(l/16) + j (j<8), 16 + 8*(l/16) + (j-8) (j>=8).
    v16h bw_ih0[4], bw_hh0[4], bw_cat1[4];
    #pragma unroll
    for (int gt = 0; gt < 4; ++gt) {
        const int n = gt * 16 + lm;   // gate column (0..63)
        v16h t0, t1, t2;
        #pragma unroll
        for (int j = 0; j < 16; ++j) {
            const int k = (j < 8) ? (half * 8 + j) : (16 + half * 8 + (j - 8));
            // layer 0 input weights: full K=32
            t0[j] = (_Float16)Wih0[n * DIN + k];
            // layer 0 recurrent weights: K 0-15 valid, 16-31 zero-padded
            t1[j] = (k < HH) ? (_Float16)Whh0[n * HH + k] : (_Float16)0.0f;
            // layer 1 concatenated weights: K 0-15 -> Wih1, K 16-31 -> Whh1
            t2[j] = (k >= HH) ? (_Float16)Whh1[n * HH + (k - HH)] : (_Float16)Wih1[n * HH + k];
        }
        bw_ih0[gt] = t0; bw_hh0[gt] = t1; bw_cat1[gt] = t2;
    }

    float bias0[4], bias1[4];
    #pragma unroll
    for (int gt = 0; gt < 4; ++gt) {
        bias0[gt] = bih0[gt * 16 + lm] + bhh0[gt * 16 + lm];
        bias1[gt] = bih1[gt * 16 + lm] + bhh1[gt * 16 + lm];
    }

    // Cell/hidden state. C/D layout: VGPR r, lane l -> M = r + 8*(l/16), N = l%16.
    v8f c1, c2, h2reg;
    #pragma unroll
    for (int r = 0; r < 8; ++r) { c1[r] = 0.0f; c2[r] = 0.0f; h2reg[r] = 0.0f; }
    // Per-lane A-layout slices of h1/h2 (row m, cols half*8 .. half*8+7).
    v8h h1slice, h2slice;
    #pragma unroll
    for (int e = 0; e < 8; ++e) { h1slice[e] = (_Float16)0.0f; h2slice[e] = (_Float16)0.0f; }

    const int m = lm;  // A-layout: this lane supplies row M = lane%16
    const float* xrow = x + (size_t)(b0 + m) * TT * DIN;

    for (int t = 0; t < TT; ++t) {
        // -------- x_t A-operand (16x32 f16). Element e, lane-half h:
        //          K = (e%8) + 8*(2*(e/8) + h).
        const float* xp = xrow + (size_t)t * DIN;
        __builtin_prefetch(xp + 8 * DIN, 0, 1);   // global_prefetch ~8 steps ahead
        v16h a_x;
        #pragma unroll
        for (int r = 0; r < 8; ++r) {
            a_x[r]     = (_Float16)xp[8 * half + r];        // K = 8*half + r
            a_x[8 + r] = (_Float16)xp[16 + 8 * half + r];   // K = 16 + 8*half + r
        }
        // h1 A-operand: elements 0-7 = h1 slice (K<16), 8-15 zero (K pad).
        v16h a_h1;
        #pragma unroll
        for (int e = 0; e < 8; ++e) { a_h1[e] = h1slice[e]; a_h1[8 + e] = (_Float16)0.0f; }

        // -------- layer 0: gates = bias0 + x_t*Wih0^T + h1*Whh0^T --------
        v8f g0[4];
        #pragma unroll
        for (int gt = 0; gt < 4; ++gt) {
            v8f acc;
            #pragma unroll
            for (int r = 0; r < 8; ++r) acc[r] = bias0[gt];
            acc = __builtin_amdgcn_wmma_f32_16x16x32_f16(false, a_x,  false, bw_ih0[gt],
                                                         (short)0, acc, false, false);
            g0[gt] = __builtin_amdgcn_wmma_f32_16x16x32_f16(false, a_h1, false, bw_hh0[gt],
                                                            (short)0, acc, false, false);
        }

        // -------- layer 0 cell update (C/D layout, pure VALU/TRANS) --------
        #pragma unroll
        for (int r = 0; r < 8; ++r) {
            const float i_ = sigmoid_fast(g0[0][r]);
            const float f_ = sigmoid_fast(g0[1][r]);
            const float gg = tanh_fast   (g0[2][r]);
            const float o_ = sigmoid_fast(g0[3][r]);
            const float c  = f_ * c1[r] + i_ * gg;
            c1[r] = c;
            const float h  = o_ * tanh_fast(c);
            lds_h1[(r + 8 * half) * 16 + lm] = (_Float16)h;   // C-layout -> row-major LDS
        }
        __syncthreads();
        h1slice = *reinterpret_cast<const v8h*>(&lds_h1[m * 16 + half * 8]);

        // -------- layer 1, K-packed: gates = bias1 + [h1|h2] * [Wih1;Whh1]^T --------
        v16h a_cat;
        #pragma unroll
        for (int e = 0; e < 8; ++e) { a_cat[e] = h1slice[e]; a_cat[8 + e] = h2slice[e]; }

        v8f g1[4];
        #pragma unroll
        for (int gt = 0; gt < 4; ++gt) {
            v8f acc;
            #pragma unroll
            for (int r = 0; r < 8; ++r) acc[r] = bias1[gt];
            g1[gt] = __builtin_amdgcn_wmma_f32_16x16x32_f16(false, a_cat, false, bw_cat1[gt],
                                                            (short)0, acc, false, false);
        }

        #pragma unroll
        for (int r = 0; r < 8; ++r) {
            const float i_ = sigmoid_fast(g1[0][r]);
            const float f_ = sigmoid_fast(g1[1][r]);
            const float gg = tanh_fast   (g1[2][r]);
            const float o_ = sigmoid_fast(g1[3][r]);
            const float c  = f_ * c2[r] + i_ * gg;
            c2[r] = c;
            const float h  = o_ * tanh_fast(c);
            h2reg[r] = h;
            lds_h2[(r + 8 * half) * 16 + lm] = (_Float16)h;
        }
        __syncthreads();
        h2slice = *reinterpret_cast<const v8h*>(&lds_h2[m * 16 + half * 8]);
    }

    // ---------------- MLP head on final h2 ----------------
    #pragma unroll
    for (int r = 0; r < 8; ++r)
        lds_last[(r + 8 * half) * 16 + lm] = h2reg[r];
    __syncthreads();

    if (lane < 16) {
        float acc = fc2b[0];
        #pragma unroll
        for (int j = 0; j < 8; ++j) {
            float s = fc1b[j];
            #pragma unroll
            for (int n = 0; n < 16; ++n)
                s += fc1w[j * 16 + n] * lds_last[lane * 16 + n];
            acc += fc2w[j] * fmaxf(s, 0.0f);
        }
        out[b0 + lane] = acc;
    }
}

extern "C" void kernel_launch(void* const* d_in, const int* in_sizes, int n_in,
                              void* d_out, int out_size, void* d_ws, size_t ws_size,
                              hipStream_t stream)
{
    (void)in_sizes; (void)n_in; (void)out_size; (void)d_ws; (void)ws_size;
    const float* x    = (const float*)d_in[0];
    const float* Wih0 = (const float*)d_in[1];
    const float* Whh0 = (const float*)d_in[2];
    const float* bih0 = (const float*)d_in[3];
    const float* bhh0 = (const float*)d_in[4];
    const float* Wih1 = (const float*)d_in[5];
    const float* Whh1 = (const float*)d_in[6];
    const float* bih1 = (const float*)d_in[7];
    const float* bhh1 = (const float*)d_in[8];
    const float* fc1w = (const float*)d_in[9];
    const float* fc1b = (const float*)d_in[10];
    const float* fc2w = (const float*)d_in[11];
    const float* fc2b = (const float*)d_in[12];
    float* out = (float*)d_out;

    dim3 grid(NB / 16);   // 256 batch tiles, one wave32 each
    dim3 block(32);
    lstm2_fused_kernel<<<grid, block, 0, stream>>>(
        x, Wih0, Whh0, bih0, bhh0, Wih1, Whh1, bih1, bhh1,
        fc1w, fc1b, fc2w, fc2b, out);
}
